// MPNNNet_83906481094707
// MI455X (gfx1250) — compile-verified, matrix operand order checked by reading, not run on hardware
//
#include <hip/hip_runtime.h>
#include <hip/hip_bf16.h>

// ---------------------------------------------------------------------------
// MPNN (NNConv + GRU, Set2Set) for MI455X / gfx1250, wave32.
//   N=20000 nodes, E=100000 edges, F_IN=16, DIM=64, HID=128, G=128 graphs
//   MP_STEPS=4, S2S_STEPS=8
// Heavy lifting:
//  * edge-net GEMM [E,128]x[128,4096] in bf16 via v_wmma_f32_16x16x32_bf16,
//    materialized once (per-edge weights are step-invariant). B operand is
//    pre-transposed so every fragment is two aligned global_load_b128.
//  * message passing streams the 800MB bf16 ew buffer 4x (bandwidth bound:
//    ~4GB @ 23.3TB/s ~= 170us) with fully coalesced b32-per-lane reads.
// ---------------------------------------------------------------------------

#define Nn 20000
#define Ee 100000
#define F_IN 16
#define DIM 64
#define HID 128
#define NG 128
#define MP_STEPS 4
#define S2S_STEPS 8

typedef __attribute__((ext_vector_type(16))) __bf16 v16bf;
typedef __attribute__((ext_vector_type(8)))  __bf16 v8bf;   // 16B vector load
typedef __attribute__((ext_vector_type(8)))  float  v8f;

// ---- bf16 helpers (explicit bit manipulation; RNE) -------------------------
__device__ __forceinline__ float bf2f(__bf16 x) {
    unsigned short u = __builtin_bit_cast(unsigned short, x);
    unsigned int v = ((unsigned int)u) << 16;
    return __builtin_bit_cast(float, v);
}
__device__ __forceinline__ float bf2f_u16(unsigned short u) {
    unsigned int v = ((unsigned int)u) << 16;
    return __builtin_bit_cast(float, v);
}
__device__ __forceinline__ __bf16 f2bf(float f) {
    unsigned int u = __builtin_bit_cast(unsigned int, f);
    unsigned int r = u + 0x7FFFu + ((u >> 16) & 1u);
    unsigned short h = (unsigned short)(r >> 16);
    return __builtin_bit_cast(__bf16, h);
}
__device__ __forceinline__ float sigf(float x) { return 1.0f / (1.0f + __expf(-x)); }

__device__ __forceinline__ void atomicMaxFloat(float* addr, float val) {
    int* ai = (int*)addr;
    int old = __float_as_int(*addr);
    while (__int_as_float(old) < val) {
        int assumed = old;
        old = atomicCAS(ai, assumed, __float_as_int(val));
        if (old == assumed) break;
    }
}

// ---- generic zero ----------------------------------------------------------
__global__ void mpnn_zero_f32(float* __restrict__ p, int n) {
    int i = blockIdx.x * blockDim.x + threadIdx.x;
    if (i < n) p[i] = 0.0f;
}

// ---- lin0: out = relu(x @ W0 + b0), [N,16]x[16,64] -------------------------
__global__ void mpnn_lin0(const float* __restrict__ x, const float* __restrict__ w,
                          const float* __restrict__ b, float* __restrict__ h,
                          __bf16* __restrict__ outb) {
    int idx = blockIdx.x * blockDim.x + threadIdx.x;           // n*64 + j
    if (idx >= Nn * DIM) return;
    int n = idx >> 6, j = idx & 63;
    const float* xr = x + n * F_IN;
    float acc = b[j];
    #pragma unroll
    for (int k = 0; k < F_IN; ++k) acc += xr[k] * w[k * DIM + j];
    acc = fmaxf(acc, 0.0f);
    h[idx] = acc;
    outb[idx] = f2bf(acc);
}

// ---- edge hidden: h1 = relu(ea @ W1 + b1), [E,2]x[2,128], store bf16 -------
__global__ void mpnn_edge_hidden(const float* __restrict__ ea, const float* __restrict__ w1,
                                 const float* __restrict__ b1, __bf16* __restrict__ h1b) {
    int idx = blockIdx.x * blockDim.x + threadIdx.x;           // e*128 + k
    if (idx >= Ee * HID) return;
    int e = idx >> 7, k = idx & 127;
    float v = b1[k] + ea[e * 2 + 0] * w1[k] + ea[e * 2 + 1] * w1[HID + k];
    h1b[idx] = f2bf(fmaxf(v, 0.0f));
}

// ---- cast + transpose nn2_w -> bf16 column-major: w2t[j*128+k] = w2[k*4096+j]
__global__ void mpnn_cast_transpose_w2(const float* __restrict__ w, __bf16* __restrict__ wt) {
    int idx = blockIdx.x * blockDim.x + threadIdx.x;           // j*128 + k
    if (idx >= HID * DIM * DIM) return;
    int j = idx >> 7, k = idx & 127;
    wt[idx] = f2bf(w[(size_t)k * (DIM * DIM) + j]);
}

// ---- degree ----------------------------------------------------------------
__global__ void mpnn_deg(const int* __restrict__ dst, float* __restrict__ deg) {
    int e = blockIdx.x * blockDim.x + threadIdx.x;
    if (e < Ee) atomicAdd(&deg[dst[e]], 1.0f);
}

// ---- ew GEMM: ew[E,4096] = h1[E,128] @ W2[128,4096] + b2, bf16 WMMA --------
// grid = (E/16, 4096/128), block = 256 (8 waves). Wave w: 16x16 tile at
// cols blockIdx.y*128 + w*16. Four v_wmma_f32_16x16x32_bf16 over K=128.
// Both fragments are built from two aligned 16B global_load_b128 each.
__global__ __launch_bounds__(256) void mpnn_ew_gemm(
    const __bf16* __restrict__ h1, const __bf16* __restrict__ w2t,
    const float* __restrict__ b2, __bf16* __restrict__ ew) {
    const int lane = threadIdx.x & 31;
    const int wave = threadIdx.x >> 5;
    const int m    = lane & 15;          // A row / B col within tile
    const int half = lane >> 4;          // K-phase selector per ISA 7.12.2
    const int rowBase = blockIdx.x * 16;
    const int colBase = blockIdx.y * 128 + wave * 16;
    const int col  = colBase + m;

    // A: 16x32 bf16 frag. lane L: row=L&15; elems[0..7]=K kb+0..7,
    // elems[8..15]=K kb+16..23, kb=(L>>4)*8 + k0.
    const __bf16* arow = h1  + (size_t)(rowBase + m) * HID + half * 8;
    // B: 32x16 frag, column-per-lane mirror; w2t is column-major so the
    // K runs are contiguous (16B aligned).
    const __bf16* bcol = w2t + (size_t)col * HID + half * 8;

    v8f c = {};
    #pragma unroll
    for (int k0 = 0; k0 < HID; k0 += 32) {
        v8bf alo = *(const v8bf*)(arow + k0);
        v8bf ahi = *(const v8bf*)(arow + k0 + 16);
        v8bf blo = *(const v8bf*)(bcol + k0);
        v8bf bhi = *(const v8bf*)(bcol + k0 + 16);
        v16bf a = __builtin_shufflevector(alo, ahi,
                    0,1,2,3,4,5,6,7,8,9,10,11,12,13,14,15);
        v16bf b = __builtin_shufflevector(blo, bhi,
                    0,1,2,3,4,5,6,7,8,9,10,11,12,13,14,15);
        c = __builtin_amdgcn_wmma_f32_16x16x32_bf16(
                false, a, false, b, (short)0, c, false, false);
    }
    // D layout: VGPR r -> M = r + half*8, N = lane&15
    const float bias = b2[col];
    #pragma unroll
    for (int r = 0; r < 8; ++r) {
        int row = rowBase + r + half * 8;
        ew[(size_t)row * 4096 + col] = f2bf(c[r] + bias);
    }
}

// ---- message: msg[e,o] = sum_d out[src[e],d] * ew[e,d,o]; scatter-add ------
// One wave32 per edge; lane owns output pair o = {2*lane, 2*lane+1} so each
// k-step is one coalesced b32 load per lane (wave reads a full 128B row).
__global__ __launch_bounds__(256) void mpnn_message(
    const int* __restrict__ src, const int* __restrict__ dst,
    const __bf16* __restrict__ ew, const __bf16* __restrict__ xb,
    float* __restrict__ agg) {
    const int lane = threadIdx.x & 31;
    const int e = blockIdx.x * 8 + (threadIdx.x >> 5);
    if (e >= Ee) return;
    const int s = src[e], d = dst[e];
    const __bf16* W  = ew + (size_t)e * 4096 + 2 * lane;   // 4B aligned per lane
    const __bf16* xs = xb + (size_t)s * DIM;
    float a0 = 0.0f, a1 = 0.0f;
    #pragma unroll
    for (int kk = 0; kk < DIM; kk += 8) {
        v8bf xv = *(const v8bf*)(xs + kk);                 // one b128, wave-uniform
        #pragma unroll
        for (int i = 0; i < 8; ++i) {
            float xf = bf2f(xv[i]);
            unsigned pw = *(const unsigned*)(W + (size_t)(kk + i) * DIM);
            a0 += xf * bf2f_u16((unsigned short)(pw & 0xFFFFu));
            a1 += xf * bf2f_u16((unsigned short)(pw >> 16));
        }
    }
    atomicAdd(&agg[(size_t)d * DIM + 2 * lane],     a0);
    atomicAdd(&agg[(size_t)d * DIM + 2 * lane + 1], a1);
}

// ---- GRU node update (grid = N/4 exact, 4 nodes x 64 threads) --------------
__global__ __launch_bounds__(256) void mpnn_gru(
    const float* __restrict__ agg, const float* __restrict__ deg,
    const float* __restrict__ convb, const float* __restrict__ wih,
    const float* __restrict__ whh, const float* __restrict__ bih,
    const float* __restrict__ bhh, float* __restrict__ h, __bf16* __restrict__ outb) {
    __shared__ float sm[4][DIM], sh[4][DIM];
    const int g = threadIdx.x >> 6, t = threadIdx.x & 63;
    const int n = blockIdx.x * 4 + g;                  // N % 4 == 0
    const float dg = fmaxf(deg[n], 1.0f);
    const float m  = fmaxf(agg[n * DIM + t] / dg + convb[t], 0.0f);
    const float hv = h[n * DIM + t];
    sm[g][t] = m; sh[g][t] = hv;
    __syncthreads();
    float gir = bih[t], giz = bih[DIM + t], gin = bih[2 * DIM + t];
    float ghr = bhh[t], ghz = bhh[DIM + t], ghn = bhh[2 * DIM + t];
    #pragma unroll 4
    for (int k = 0; k < DIM; ++k) {
        float mk = sm[g][k], hk = sh[g][k];
        gir += mk * wih[k * 192 + t];       ghr += hk * whh[k * 192 + t];
        giz += mk * wih[k * 192 + 64 + t];  ghz += hk * whh[k * 192 + 64 + t];
        gin += mk * wih[k * 192 + 128 + t]; ghn += hk * whh[k * 192 + 128 + t];
    }
    float r  = sigf(gir + ghr);
    float z  = sigf(giz + ghz);
    float ng = tanhf(gin + r * ghn);
    float hn = (1.0f - z) * ng + z * hv;
    h[n * DIM + t] = hn;
    outb[n * DIM + t] = f2bf(hn);
}

// ---- Set2Set LSTM step (one block per graph) + reset of attn scratch -------
__global__ __launch_bounds__(256) void mpnn_lstm(
    const float* __restrict__ wih, const float* __restrict__ whh,
    const float* __restrict__ bih, const float* __restrict__ bhh,
    const float* __restrict__ qs, float* __restrict__ hl, float* __restrict__ cl,
    float* __restrict__ mg, float* __restrict__ sg, float* __restrict__ rbuf) {
    const int g = blockIdx.x, j = threadIdx.x;         // 256 threads
    __shared__ float q[2 * DIM], hh[DIM], gates[4 * DIM];
    if (j < 2 * DIM) q[j]  = qs[g * 2 * DIM + j];
    if (j < DIM)     hh[j] = hl[g * DIM + j];
    __syncthreads();
    float acc = bih[j] + bhh[j];
    #pragma unroll 4
    for (int k = 0; k < 2 * DIM; ++k) acc += q[k] * wih[k * 256 + j];
    #pragma unroll 4
    for (int k = 0; k < DIM; ++k)     acc += hh[k] * whh[k * 256 + j];
    gates[j] = acc;
    __syncthreads();
    if (j < DIM) {
        float ii = sigf(gates[j]);
        float ff = sigf(gates[DIM + j]);
        float gg = tanhf(gates[2 * DIM + j]);
        float oo = sigf(gates[3 * DIM + j]);
        float c  = ff * cl[g * DIM + j] + ii * gg;
        cl[g * DIM + j] = c;
        hl[g * DIM + j] = oo * tanhf(c);
        rbuf[g * DIM + j] = 0.0f;
    }
    if (j == DIM) { mg[g] = -3.0e38f; sg[g] = 0.0f; }
}

// ---- attention score e[n] = <out[n], q[batch[n]]> + segment max ------------
__global__ __launch_bounds__(256) void mpnn_attn_e(
    const float* __restrict__ h, const int* __restrict__ batch,
    const float* __restrict__ hl, float* __restrict__ ebuf, float* __restrict__ mg) {
    const int lane = threadIdx.x & 31;
    const int n = blockIdx.x * 8 + (threadIdx.x >> 5);
    if (n >= Nn) return;
    const int b = batch[n];
    const float* q = hl + b * DIM;
    const float* o = h + (size_t)n * DIM;
    float p = o[lane] * q[lane] + o[lane + 32] * q[lane + 32];
    #pragma unroll
    for (int s = 16; s > 0; s >>= 1) p += __shfl_xor(p, s, 32);
    if (lane == 0) { ebuf[n] = p; atomicMaxFloat(&mg[b], p); }
}

// ---- exp + segment sum -----------------------------------------------------
__global__ void mpnn_attn_exp(const float* __restrict__ ebuf, const int* __restrict__ batch,
                              const float* __restrict__ mg, float* __restrict__ exb,
                              float* __restrict__ sg) {
    int n = blockIdx.x * blockDim.x + threadIdx.x;
    if (n >= Nn) return;
    int b = batch[n];
    float ex = __expf(ebuf[n] - mg[b]);
    exb[n] = ex;
    atomicAdd(&sg[b], ex);
}

// ---- weighted readout r = segment_sum(a * out) -----------------------------
__global__ __launch_bounds__(256) void mpnn_attn_r(
    const float* __restrict__ h, const int* __restrict__ batch,
    const float* __restrict__ exb, const float* __restrict__ sg,
    float* __restrict__ rbuf) {
    const int lane = threadIdx.x & 31;
    const int n = blockIdx.x * 8 + (threadIdx.x >> 5);
    if (n >= Nn) return;
    const int b = batch[n];
    const float a = exb[n] / sg[b];
    atomicAdd(&rbuf[b * DIM + lane],      a * h[(size_t)n * DIM + lane]);
    atomicAdd(&rbuf[b * DIM + 32 + lane], a * h[(size_t)n * DIM + 32 + lane]);
}

// ---- q_star = concat(q, r) -------------------------------------------------
__global__ void mpnn_qstar(const float* __restrict__ hl, const float* __restrict__ rbuf,
                           float* __restrict__ qs) {
    int g = blockIdx.x, j = threadIdx.x;               // 128 threads
    qs[g * 2 * DIM + j] = (j < DIM) ? hl[g * DIM + j] : rbuf[g * DIM + (j - DIM)];
}

// ---- head: y = relu(qs@W1+b1); logits = y@W2+b2; log_softmax ---------------
__global__ __launch_bounds__(64) void mpnn_head(
    const float* __restrict__ qs, const float* __restrict__ w1,
    const float* __restrict__ b1, const float* __restrict__ w2,
    const float* __restrict__ b2, float* __restrict__ out) {
    const int g = blockIdx.x, j = threadIdx.x;         // 64 threads
    __shared__ float q[2 * DIM], y[DIM], lg[10];
    q[j] = qs[g * 2 * DIM + j];
    q[DIM + j] = qs[g * 2 * DIM + DIM + j];
    __syncthreads();
    float acc = b1[j];
    #pragma unroll 4
    for (int k = 0; k < 2 * DIM; ++k) acc += q[k] * w1[k * DIM + j];
    y[j] = fmaxf(acc, 0.0f);
    __syncthreads();
    if (j < 10) {
        float a = b2[j];
        #pragma unroll 4
        for (int k = 0; k < DIM; ++k) a += y[k] * w2[k * 10 + j];
        lg[j] = a;
    }
    __syncthreads();
    if (j == 0) {
        float m = lg[0];
        for (int i = 1; i < 10; ++i) m = fmaxf(m, lg[i]);
        float s = 0.0f;
        for (int i = 0; i < 10; ++i) s += __expf(lg[i] - m);
        float lse = m + __logf(s);
        for (int i = 0; i < 10; ++i) out[g * 10 + i] = lg[i] - lse;
    }
}

// ---------------------------------------------------------------------------
extern "C" void kernel_launch(void* const* d_in, const int* in_sizes, int n_in,
                              void* d_out, int out_size, void* d_ws, size_t ws_size,
                              hipStream_t stream) {
    (void)in_sizes; (void)n_in; (void)out_size; (void)ws_size;
    const float* x       = (const float*)d_in[0];
    const int*   ei      = (const int*)d_in[1];
    const float* ea      = (const float*)d_in[2];
    const int*   batch   = (const int*)d_in[3];
    const float* lin0_w  = (const float*)d_in[4];
    const float* lin0_b  = (const float*)d_in[5];
    const float* nn1_w   = (const float*)d_in[6];
    const float* nn1_b   = (const float*)d_in[7];
    const float* nn2_w   = (const float*)d_in[8];
    const float* nn2_b   = (const float*)d_in[9];
    const float* conv_b  = (const float*)d_in[10];
    const float* gruWih  = (const float*)d_in[11];
    const float* gruWhh  = (const float*)d_in[12];
    const float* gruBih  = (const float*)d_in[13];
    const float* gruBhh  = (const float*)d_in[14];
    const float* lstmWih = (const float*)d_in[15];
    const float* lstmWhh = (const float*)d_in[16];
    const float* lstmBih = (const float*)d_in[17];
    const float* lstmBhh = (const float*)d_in[18];
    const float* lin1_w  = (const float*)d_in[19];
    const float* lin1_b  = (const float*)d_in[20];
    const float* lin2_w  = (const float*)d_in[21];
    const float* lin2_b  = (const float*)d_in[22];
    float* out = (float*)d_out;

    const int* src = ei;
    const int* dst = ei + Ee;

    // ---- carve workspace (256B aligned); total ~870 MB --------------------
    char* p = (char*)d_ws;
    auto carve = [&](size_t bytes) -> char* {
        char* r = p; p += (bytes + 255) & ~(size_t)255; return r;
    };
    float*  h    = (float*)carve((size_t)Nn * DIM * 4);
    __bf16* outb = (__bf16*)carve((size_t)Nn * DIM * 2);
    __bf16* h1b  = (__bf16*)carve((size_t)Ee * HID * 2);
    __bf16* w2t  = (__bf16*)carve((size_t)HID * DIM * DIM * 2);
    __bf16* ewb  = (__bf16*)carve((size_t)Ee * DIM * DIM * 2);   // 800 MB
    float*  agg  = (float*)carve((size_t)Nn * DIM * 4);
    float*  deg  = (float*)carve((size_t)Nn * 4);
    float*  ebuf = (float*)carve((size_t)Nn * 4);
    float*  exb  = (float*)carve((size_t)Nn * 4);
    float*  qs   = (float*)carve((size_t)NG * 2 * DIM * 4);
    float*  hl   = (float*)carve((size_t)NG * DIM * 4);
    float*  cl   = (float*)carve((size_t)NG * DIM * 4);
    float*  mg   = (float*)carve((size_t)NG * 4);
    float*  sg   = (float*)carve((size_t)NG * 4);
    float*  rbuf = (float*)carve((size_t)NG * DIM * 4);

    // ---- encoder + edge network ------------------------------------------
    mpnn_lin0<<<(Nn * DIM + 255) / 256, 256, 0, stream>>>(x, lin0_w, lin0_b, h, outb);
    mpnn_edge_hidden<<<(Ee * HID + 255) / 256, 256, 0, stream>>>(ea, nn1_w, nn1_b, h1b);
    mpnn_cast_transpose_w2<<<(HID * DIM * DIM + 255) / 256, 256, 0, stream>>>(nn2_w, w2t);
    mpnn_zero_f32<<<(Nn + 255) / 256, 256, 0, stream>>>(deg, Nn);
    mpnn_deg<<<(Ee + 255) / 256, 256, 0, stream>>>(dst, deg);
    // big WMMA GEMM: [E,128] x [128,4096] -> ew (bf16)
    mpnn_ew_gemm<<<dim3(Ee / 16, (DIM * DIM) / 128), 256, 0, stream>>>(h1b, w2t, nn2_b, ewb);

    // ---- message passing --------------------------------------------------
    for (int s = 0; s < MP_STEPS; ++s) {
        mpnn_zero_f32<<<(Nn * DIM + 255) / 256, 256, 0, stream>>>(agg, Nn * DIM);
        mpnn_message<<<Ee / 8, 256, 0, stream>>>(src, dst, ewb, outb, agg);
        mpnn_gru<<<Nn / 4, 256, 0, stream>>>(agg, deg, conv_b, gruWih, gruWhh,
                                             gruBih, gruBhh, h, outb);
    }

    // ---- Set2Set ----------------------------------------------------------
    mpnn_zero_f32<<<(NG * 2 * DIM + 255) / 256, 256, 0, stream>>>(qs, NG * 2 * DIM);
    mpnn_zero_f32<<<(NG * DIM + 255) / 256, 256, 0, stream>>>(hl, NG * DIM);
    mpnn_zero_f32<<<(NG * DIM + 255) / 256, 256, 0, stream>>>(cl, NG * DIM);
    for (int s = 0; s < S2S_STEPS; ++s) {
        mpnn_lstm<<<NG, 256, 0, stream>>>(lstmWih, lstmWhh, lstmBih, lstmBhh,
                                          qs, hl, cl, mg, sg, rbuf);
        mpnn_attn_e<<<Nn / 8, 256, 0, stream>>>(h, batch, hl, ebuf, mg);
        mpnn_attn_exp<<<(Nn + 255) / 256, 256, 0, stream>>>(ebuf, batch, mg, exb, sg);
        mpnn_attn_r<<<Nn / 8, 256, 0, stream>>>(h, batch, exb, sg, rbuf);
        mpnn_qstar<<<NG, 128, 0, stream>>>(hl, rbuf, qs);
    }

    // ---- head -------------------------------------------------------------
    mpnn_head<<<NG, 64, 0, stream>>>(qs, lin1_w, lin1_b, lin2_w, lin2_b, out);
}
